// LePEAttention_446676599473
// MI455X (gfx1250) — compile-verified, matrix operand order checked by reading, not run on hardware
//
#include <hip/hip_runtime.h>
#include <hip/hip_bf16.h>

typedef __attribute__((ext_vector_type(16))) _Float16 v16h;
typedef __attribute__((ext_vector_type(8)))  _Float16 v8h;
typedef __attribute__((ext_vector_type(8)))  float    v8f;

#define BB    32
#define IMG_W 56
#define H_SP  56
#define W_SP  7
#define CC    96
#define NHEAD 4
#define DD    24
#define LL    (56 * 56)
#define WIN   392          // tokens per window (56*7)
#define NWIN  8            // windows per image
#define MT    25           // 25 M/N tiles of 16 -> 400 padded rows
#define RPAD  416          // padded token dim for PV K-loop (13*32)

// Load a 16x32 f16 operand row into the A/B per-lane layout:
// lane (m or n) = lane&15, half = lane>>4 owns K runs [half*8, half*8+8) and
// [16+half*8, 16+half*8+8)  (two contiguous 16B chunks -> ds_load_b128 x2).
static __device__ __forceinline__ v16h ld_ab(const _Float16* row, int half) {
    const v8h* p0 = (const v8h*)(row + half * 8);
    const v8h* p1 = (const v8h*)(row + 16 + half * 8);
    v8h lo = *p0;
    v8h hi = *p1;
    v16h r;
#pragma unroll
    for (int i = 0; i < 8; ++i) { r[i] = lo[i]; r[i + 8] = hi[i]; }
    return r;
}

__global__ __launch_bounds__(256) void lepe_attn_kernel(
    const float* __restrict__ qkv,     // (3, B, L, C)
    const float* __restrict__ w_lepe,  // (C, 1, 3, 3)
    const float* __restrict__ b_lepe,  // (C,)
    float* __restrict__ out)           // (B, L, C)
{
    extern __shared__ _Float16 smem[];
    _Float16* sQ    = smem;              // 416 rows x 32 cols (f16)
    _Float16* sK    = smem + 13312;      // 416 x 32
    _Float16* sVT   = smem + 26624;      // 32 rows (d) x 416 cols (t)
    _Float16* sPall = smem + 39936;      // 8 waves x (16 x 32) staging

    const int nh  = blockIdx.x & 3;
    const int win = blockIdx.x >> 2;
    const int b   = win >> 3;
    const int wi  = win & 7;

    const int tid = threadIdx.x;

    // ---- zero-fill padded LDS (Q, K, VT) ----
    uint32_t* z = (uint32_t*)smem;
    for (int i = tid; i < 19968; i += 256) z[i] = 0u;
    __syncthreads();

    // ---- stage Q (scaled), K, V^T into LDS as f16 ----
    const size_t BLC   = (size_t)BB * LL * CC;
    const size_t base  = (size_t)b * LL * CC + (size_t)nh * DD;
    const float  scale = 0.2041241452319315f;  // 24^-0.5
    for (int e = tid; e < WIN * DD; e += 256) {
        int t  = e / DD;
        int d  = e - t * DD;
        int hh = t / W_SP;
        int ww = t - hh * W_SP;
        size_t gi = base + (size_t)(hh * IMG_W + wi * W_SP + ww) * CC + d;
        float qv = qkv[gi];
        float kv = qkv[BLC + gi];
        float vv = qkv[2 * BLC + gi];
        sQ[t * 32 + d]    = (_Float16)(qv * scale);
        sK[t * 32 + d]    = (_Float16)kv;
        sVT[d * RPAD + t] = (_Float16)vv;
    }
    __syncthreads();

    const int wave = tid >> 5;
    const int lane = tid & 31;
    const int lm   = lane & 15;
    const int half = lane >> 4;
    _Float16* sP = sPall + wave * 512;

    const size_t obase = (size_t)b * LL * CC;

    for (int mt = wave; mt < MT; mt += 8) {
        // ---------------- S = (Q*scale) . K^T  (one 16x400 row-block) --------
        v16h aq = ld_ab(sQ + (mt * 16 + lm) * 32, half);
        v8f s[25];
#pragma unroll
        for (int j = 0; j < 25; ++j) {
            v16h bk = ld_ab(sK + (j * 16 + lm) * 32, half);
            v8f c0 = {};
            s[j] = __builtin_amdgcn_wmma_f32_16x16x32_f16(
                false, aq, false, bk, (short)0, c0, false, false);
        }
        // mask padded columns 392..399 (tile 24, n = lm >= 8)
        if (lm >= 8) {
#pragma unroll
            for (int r = 0; r < 8; ++r) s[24][r] = -3.0e38f;
        }

        // ---------------- softmax over each row (16 lanes per half) ---------
#pragma unroll
        for (int r = 0; r < 8; ++r) {
            float mx = -3.4e38f;
#pragma unroll
            for (int j = 0; j < 25; ++j) mx = fmaxf(mx, s[j][r]);
#pragma unroll
            for (int m = 1; m <= 8; m <<= 1) mx = fmaxf(mx, __shfl_xor(mx, m, 32));
            float sum = 0.0f;
#pragma unroll
            for (int j = 0; j < 25; ++j) {
                float e = __expf(s[j][r] - mx);
                s[j][r] = e;
                sum += e;
            }
#pragma unroll
            for (int m = 1; m <= 8; m <<= 1) sum += __shfl_xor(sum, m, 32);
            float inv = 1.0f / sum;
#pragma unroll
            for (int j = 0; j < 25; ++j) s[j][r] *= inv;
        }

        // ---------------- X = P . V  (K over 416 padded tokens) -------------
        v8f x[2];
        x[0] = (v8f){};
        x[1] = (v8f){};
#pragma unroll
        for (int kt = 0; kt < 13; ++kt) {
            // scatter this 16x32 P chunk (C/D layout) into per-wave staging
#pragma unroll
            for (int jj = 0; jj < 2; ++jj) {
                const int j = 2 * kt + jj;
                if (j < 25) {
#pragma unroll
                    for (int r = 0; r < 8; ++r)
                        sP[(r + 8 * half) * 32 + jj * 16 + lm] = (_Float16)s[j][r];
                } else {
#pragma unroll
                    for (int r = 0; r < 8; ++r)
                        sP[(r + 8 * half) * 32 + jj * 16 + lm] = (_Float16)0.0f;
                }
            }
            // same-wave LDS RAW across lanes: drain DS counter
            asm volatile("s_wait_dscnt 0" ::: "memory");
            v16h ap  = ld_ab(sP + lm * 32, half);
            v16h bv0 = ld_ab(sVT + lm * RPAD + kt * 32, half);
            v16h bv1 = ld_ab(sVT + (16 + lm) * RPAD + kt * 32, half);
            x[0] = __builtin_amdgcn_wmma_f32_16x16x32_f16(
                false, ap, false, bv0, (short)0, x[0], false, false);
            x[1] = __builtin_amdgcn_wmma_f32_16x16x32_f16(
                false, ap, false, bv1, (short)0, x[1], false, false);
        }

        // ---------------- epilogue: + LePE (depthwise 3x3) and store --------
#pragma unroll
        for (int nt = 0; nt < 2; ++nt) {
            const int d = nt * 16 + lm;       // channel within head
            if (d >= DD) continue;
            const int c = nh * DD + d;
            float wv[9];
#pragma unroll
            for (int i = 0; i < 9; ++i) wv[i] = w_lepe[c * 9 + i];
            const float bias = b_lepe[c];
            const _Float16* vrow = sVT + d * RPAD;
#pragma unroll
            for (int r = 0; r < 8; ++r) {
                const int t = mt * 16 + r + 8 * half;
                if (t >= WIN) continue;
                const int hh = t / W_SP;
                const int ww = t - hh * W_SP;
                float acc = bias;
#pragma unroll
                for (int dy = -1; dy <= 1; ++dy) {
                    const int y = hh + dy;
                    if (y < 0 || y >= H_SP) continue;
#pragma unroll
                    for (int dx = -1; dx <= 1; ++dx) {
                        const int xx = ww + dx;
                        if (xx < 0 || xx >= W_SP) continue;
                        acc += wv[(dy + 1) * 3 + (dx + 1)] *
                               (float)vrow[y * W_SP + xx];
                    }
                }
                out[obase + (size_t)(hh * IMG_W + wi * W_SP + ww) * CC + c] =
                    x[nt][r] + acc;
            }
        }
    }
}

extern "C" void kernel_launch(void* const* d_in, const int* in_sizes, int n_in,
                              void* d_out, int out_size, void* d_ws, size_t ws_size,
                              hipStream_t stream) {
    const float* qkv    = (const float*)d_in[0];
    const float* w_lepe = (const float*)d_in[1];
    const float* b_lepe = (const float*)d_in[2];
    float* out = (float*)d_out;

    dim3 grid(BB * NWIN * NHEAD);   // 1024 blocks: one per (window, head)
    dim3 block(256);                // 8 wave32
    size_t shmem = (size_t)44032 * sizeof(_Float16);  // 88 KB
    hipLaunchKernelGGL(lepe_attn_kernel, grid, block, shmem, stream,
                       qkv, w_lepe, b_lepe, out);
}
